// MultiHeadAttention_48619029791119
// MI455X (gfx1250) — matrix operand order repaired and checked by reading, not compiled, not run on hardware
//
#include <hip/hip_runtime.h>
#include <hip/hip_bf16.h>
#include <math.h>

typedef __attribute__((ext_vector_type(16))) __bf16 bf16x16;
typedef __attribute__((ext_vector_type(8)))  float  f32x8;
typedef __attribute__((ext_vector_type(4)))  unsigned int u32x4;
typedef __attribute__((ext_vector_type(8)))  unsigned int u32x8;

union BV { bf16x16 v; uint4 q[2]; __bf16 h[16]; };
union O8 { uint4 q; __bf16 h[8]; };
union C4 { uint2 q; __bf16 h[4]; };

__device__ inline f32x8 zero8() {
  f32x8 z = {0.f, 0.f, 0.f, 0.f, 0.f, 0.f, 0.f, 0.f};
  return z;
}

__device__ inline f32x8 wmma_bf16(bf16x16 a, bf16x16 b, f32x8 c) {
  return __builtin_amdgcn_wmma_f32_16x16x32_bf16(
      false, a, false, b, (short)0, c, false, false);
}

// A-operand (16x32 bf16): lane m=lane&15 holds row m; K 0..15 (lanes 0-15)
// or K 16..31 (lanes 16-31), contiguous halves
__device__ inline bf16x16 load_a_op(const __bf16* base, int row, int rowStride,
                                    int kBase, int lane) {
  const __bf16* p = base + row * rowStride + kBase + ((lane >> 4) << 4);
  BV u;
  u.q[0] = *(const uint4*)(p);
  u.q[1] = *(const uint4*)(p + 8);
  return u.v;
}

// B-operand (32x16 bf16): lane n=lane&15 holds column n; K 0..15 (lanes 0-15)
// or K 16..31 (lanes 16-31), contiguous halves
__device__ inline bf16x16 load_b_op(const __bf16* base, int row, int rowStride,
                                    int kBase, int lane) {
  const __bf16* p = base + row * rowStride + kBase + ((lane >> 4) << 4);
  BV u;
  u.q[0] = ((const uint4*)p)[0];
  u.q[1] = ((const uint4*)p)[1];
  return u.v;
}

// Flat/generic pointer low 32 bits == wave-relative LDS byte address (ISA 10.2)
__device__ inline unsigned int lds_addr_of(const void* p) {
  return (unsigned int)(uintptr_t)p;
}

// Per-lane async copy global -> LDS (16B/lane), tracked by ASYNCcnt
__device__ inline void async_load_b128_to_lds(void* lds, const void* g) {
  asm volatile("global_load_async_to_lds_b128 %0, %1, off"
               :: "v"(lds_addr_of(lds)),
                  "v"((unsigned long long)(uintptr_t)g)
               : "memory");
}

// TDM: one 2D tile, 128 rows x 64 bytes, global row stride 2048 B, packed into
// LDS contiguously. data_size=8B units: tile_dim0=8, tile_dim1=128,
// tensor_dim0_stride=256. Tracked by TENSORcnt.
__device__ inline void tdm_load_tile_128x64B(void* lds, const void* gptr) {
  unsigned long long ga = (unsigned long long)(uintptr_t)gptr;
  u32x4 g0;
  g0[0] = 1u;                                            // count=1, no gather
  g0[1] = lds_addr_of(lds);                              // lds_addr
  g0[2] = (unsigned int)ga;                              // global_addr[31:0]
  g0[3] = (unsigned int)((ga >> 32) & 0x01FFFFFFull)     // global_addr[56:32]
        | (2u << 30);                                    // type=2 (image)
  u32x8 g1;
  g1[0] = 0x00030000u;          // wg_mask=0, data_size=3 (8B), no pad/iterate
  g1[1] = 256u << 16;           // tensor_dim0[15:0]=256 (bits 63:48)
  g1[2] = 0u;                   // tensor_dim0[31:16]=0, tensor_dim1[15:0]=0
  g1[3] = 0x00080010u;          // tensor_dim1[31:16]=0x10 (=1M rows), tile_dim0=8
  g1[4] = 128u;                 // tile_dim1=128, tile_dim2=0
  g1[5] = 256u;                 // tensor_dim0_stride[31:0]=256 units (2048 B)
  g1[6] = 0u;                   // stride hi, tensor_dim1_stride lo
  g1[7] = 0u;
  asm volatile("tensor_load_to_lds %0, %1" :: "s"(g0), "s"(g1) : "memory");
}

// ---------------------------------------------------------------------------
// fp32 W (K x N) -> bf16 W^T (N x K)
// ---------------------------------------------------------------------------
__global__ __launch_bounds__(256)
void wt_convert_kernel(const float* __restrict__ W, __bf16* __restrict__ Wt) {
  int idx = blockIdx.x * 256 + threadIdx.x;   // 0 .. 1M-1
  int k = idx >> 10;
  int n = idx & 1023;
  Wt[(size_t)n * 1024 + k] = (__bf16)W[idx];
}

// fp32 -> bf16, same layout, 4 elements/thread
__global__ __launch_bounds__(256)
void f32_to_bf16_kernel(const float* __restrict__ X, __bf16* __restrict__ Y) {
  size_t i = ((size_t)blockIdx.x * 256 + threadIdx.x) * 4;
  float4 f = *(const float4*)(X + i);
  C4 c;
  c.h[0] = (__bf16)f.x; c.h[1] = (__bf16)f.y;
  c.h[2] = (__bf16)f.z; c.h[3] = (__bf16)f.w;
  *(uint2*)(Y + i) = c.q;
}

// ---------------------------------------------------------------------------
// 128x128-tile WMMA GEMM, K=N=1024, TDM-staged + double-buffered LDS.
// MODE 0: bf16 out head-split  dst[((b*16+h)*2048+s)*64+dk]
// MODE 1: bf16 out head-splitT dst[((b*16+h)*64+dk)*2048+s]
// MODE 2: f32 out row-major    dst[m*1024+n]
// ---------------------------------------------------------------------------
template <int MODE>
__global__ __launch_bounds__(256)
void gemm128_wmma(const __bf16* __restrict__ A, const __bf16* __restrict__ Bt,
                  const float* __restrict__ bias, void* __restrict__ dst)
{
  const int K = 1024, N = 1024;
  __shared__ __align__(16) __bf16 sA[2][128 * 32];  // 2 x 8 KB
  __shared__ __align__(16) __bf16 sB[2][128 * 32];  // 2 x 8 KB

  const int tid  = threadIdx.x;
  const int lane = tid & 31;
  const int wave = tid >> 5;
  const int m0 = blockIdx.x * 128, n0 = blockIdx.y * 128;
  const int wm = (wave >> 1) * 32;
  const int wn = (wave & 1) * 64;

  const __bf16* aRow = A  + (size_t)m0 * K;
  const __bf16* bRow = Bt + (size_t)n0 * K;

  f32x8 acc[2][4];
  #pragma unroll
  for (int i = 0; i < 2; ++i)
    #pragma unroll
    for (int j = 0; j < 4; ++j) acc[i][j] = zero8();

  if (wave == 0) tdm_load_tile_128x64B(&sA[0][0], aRow);
  if (wave == 1) tdm_load_tile_128x64B(&sB[0][0], bRow);

  const int NIT = K / 32;
  for (int it = 0; it < NIT; ++it) {
    if (it + 1 < NIT) {
      if (wave == 0) tdm_load_tile_128x64B(&sA[(it + 1) & 1][0], aRow + (it + 1) * 32);
      if (wave == 1) tdm_load_tile_128x64B(&sB[(it + 1) & 1][0], bRow + (it + 1) * 32);
      __builtin_amdgcn_s_wait_tensorcnt(1);   // oldest (current tile) done
    } else {
      __builtin_amdgcn_s_wait_tensorcnt(0);
    }
    __syncthreads();

    const __bf16* cA = sA[it & 1];
    const __bf16* cB = sB[it & 1];
    bf16x16 a0 = load_a_op(cA, wm + (lane & 15),      32, 0, lane);
    bf16x16 a1 = load_a_op(cA, wm + 16 + (lane & 15), 32, 0, lane);
    #pragma unroll
    for (int ni = 0; ni < 4; ++ni) {
      bf16x16 b = load_b_op(cB, wn + ni * 16 + (lane & 15), 32, 0, lane);
      acc[0][ni] = wmma_bf16(a0, b, acc[0][ni]);
      acc[1][ni] = wmma_bf16(a1, b, acc[1][ni]);
    }
    __syncthreads();
  }

  const int half = lane >> 4;
  #pragma unroll
  for (int mi = 0; mi < 2; ++mi) {
    #pragma unroll
    for (int ni = 0; ni < 4; ++ni) {
      int n = n0 + wn + ni * 16 + (lane & 15);
      float bvv = bias[n];
      int mbase = m0 + wm + mi * 16 + half * 8;
      #pragma unroll
      for (int r = 0; r < 8; ++r) {
        int m = mbase + r;
        float val = acc[mi][ni][r] + bvv;
        if constexpr (MODE == 2) {
          ((float*)dst)[(size_t)m * N + n] = val;
        } else {
          int bb = m >> 11, s = m & 2047;
          int hh = n >> 6,  dk = n & 63;
          __bf16* o = (__bf16*)dst;
          if constexpr (MODE == 0)
            o[(((size_t)(bb * 16 + hh) * 2048) + s) * 64 + dk] = (__bf16)val;
          else
            o[(((size_t)(bb * 16 + hh) * 64) + dk) * 2048 + s] = (__bf16)val;
        }
      }
    }
  }
}

// ---------------------------------------------------------------------------
// Fused flash attention; async-to-LDS staged, double-buffered K/V tiles.
// Scores computed transposed (keys x queries): softmax row reduce is 16
// in-register values + one shfl_xor(16). out^T = V^T * P^T.
// ---------------------------------------------------------------------------
__global__ __launch_bounds__(256)
void attn_kernel(const __bf16* __restrict__ qw, const __bf16* __restrict__ kw,
                 const __bf16* __restrict__ vtw, __bf16* __restrict__ comb)
{
  const int S = 2048, DK = 64, H = 16;
  __shared__ __align__(16) __bf16 sK[2][32 * 64];  // keys x dk
  __shared__ __align__(16) __bf16 sV[2][64 * 32];  // dk x keys

  const int tid  = threadIdx.x;
  const int lane = tid & 31;
  const int wave = tid >> 5;
  const int b = blockIdx.z, h = blockIdx.y;
  const int q = blockIdx.x * 128 + wave * 16 + (lane & 15);

  const size_t head = (size_t)(b * H + h) * S * DK;
  const __bf16* qbase = qw + head;
  const __bf16* kbase = kw + head;
  const __bf16* vbase = vtw + head;   // (dk, s) per head

  bf16x16 qB0, qB1;
  {
    const __bf16* p = qbase + (size_t)q * DK + ((lane >> 4) << 4);
    BV u0; u0.q[0] = ((const uint4*)p)[0];        u0.q[1] = ((const uint4*)p)[1];        qB0 = u0.v;
    BV u1; u1.q[0] = ((const uint4*)(p + 32))[0]; u1.q[1] = ((const uint4*)(p + 32))[1]; qB1 = u1.v;
  }

  f32x8 acc[4];
  #pragma unroll
  for (int sb = 0; sb < 4; ++sb) acc[sb] = zero8();
  float mrun = -INFINITY, lrun = 0.f;
  const float scale = 0.125f;

  const int krow = tid >> 3, kseg = tid & 7;  // 32 rows x 8 x 16B
  const int vrow = tid >> 2, vseg = tid & 3;  // 64 rows x 4 x 16B

  auto stage = [&](int buf, int kt) {
    async_load_b128_to_lds(&sK[buf][krow * 64 + kseg * 8],
                           kbase + (size_t)(kt + krow) * DK + kseg * 8);
    async_load_b128_to_lds(&sV[buf][vrow * 32 + vseg * 8],
                           vbase + (size_t)vrow * S + kt + vseg * 8);
  };

  const int NIT = S / 32;
  stage(0, 0);
  for (int it = 0; it < NIT; ++it) {
    if (it + 1 < NIT) {
      stage((it + 1) & 1, (it + 1) * 32);
      asm volatile("s_wait_asynccnt 0x2" ::: "memory");  // current tile retired
    } else {
      asm volatile("s_wait_asynccnt 0x0" ::: "memory");
    }
    __syncthreads();

    const __bf16* cK = sK[it & 1];
    const __bf16* cV = sV[it & 1];

    f32x8 st0 = zero8(), st1 = zero8();
    {
      bf16x16 aK;
      aK = load_a_op(cK, (lane & 15),      64, 0,  lane); st0 = wmma_bf16(aK, qB0, st0);
      aK = load_a_op(cK, (lane & 15),      64, 32, lane); st0 = wmma_bf16(aK, qB1, st0);
      aK = load_a_op(cK, 16 + (lane & 15), 64, 0,  lane); st1 = wmma_bf16(aK, qB0, st1);
      aK = load_a_op(cK, 16 + (lane & 15), 64, 32, lane); st1 = wmma_bf16(aK, qB1, st1);
    }

    float p0[8], p1[8];
    float mloc = -INFINITY;
    #pragma unroll
    for (int r = 0; r < 8; ++r) {
      p0[r] = st0[r] * scale;
      p1[r] = st1[r] * scale;
      mloc = fmaxf(mloc, fmaxf(p0[r], p1[r]));
    }
    mloc = fmaxf(mloc, __shfl_xor(mloc, 16, 32));
    float mnew  = fmaxf(mrun, mloc);
    float alpha = __expf(mrun - mnew);
    float lsum = 0.f;
    #pragma unroll
    for (int r = 0; r < 8; ++r) {
      p0[r] = __expf(p0[r] - mnew);
      p1[r] = __expf(p1[r] - mnew);
      lsum += p0[r] + p1[r];
    }
    lsum += __shfl_xor(lsum, 16, 32);
    lrun = lrun * alpha + lsum;
    mrun = mnew;
    #pragma unroll
    for (int sb = 0; sb < 4; ++sb)
      #pragma unroll
      for (int r = 0; r < 8; ++r) acc[sb][r] *= alpha;

    BV pb;
    const bool hi = lane >= 16;
    #pragma unroll
    for (int j = 0; j < 8; ++j) {
      float x0 = __shfl_xor(p0[j], 16, 32);
      float x1 = __shfl_xor(p1[j], 16, 32);
      pb.h[j]     = (__bf16)(hi ? x1 : p0[j]);
      pb.h[8 + j] = (__bf16)(hi ? p1[j] : x0);
    }

    #pragma unroll
    for (int sb = 0; sb < 4; ++sb) {
      bf16x16 aV = load_a_op(cV, sb * 16 + (lane & 15), 32, 0, lane);
      acc[sb] = wmma_bf16(aV, pb.v, acc[sb]);
    }
    __syncthreads();
  }

  float inv = 1.f / lrun;
  #pragma unroll
  for (int sb = 0; sb < 4; ++sb) {
    O8 o;
    #pragma unroll
    for (int r = 0; r < 8; ++r) o.h[r] = (__bf16)(acc[sb][r] * inv);
    int dkl = sb * 16 + (lane >> 4) * 8;
    __bf16* dp = comb + ((size_t)(b * S + q)) * 1024 + h * 64 + dkl;
    *(uint4*)dp = o.q;
  }
}

// ---------------------------------------------------------------------------
extern "C" void kernel_launch(void* const* d_in, const int* in_sizes, int n_in,
                              void* d_out, int out_size, void* d_ws, size_t ws_size,
                              hipStream_t stream)
{
  (void)in_sizes; (void)n_in; (void)out_size; (void)ws_size;
  const float* Q  = (const float*)d_in[0];
  const float* K  = (const float*)d_in[1];
  const float* V  = (const float*)d_in[2];
  const float* Wq = (const float*)d_in[3];
  const float* bq = (const float*)d_in[4];
  const float* Wk = (const float*)d_in[5];
  const float* bk = (const float*)d_in[6];
  const float* Wv = (const float*)d_in[7];
  const float* bv = (const float*)d_in[8];
  const float* Wo = (const float*)d_in[9];
  const float* bo = (const float*)d_in[10];
  float* out = (float*)d_out;

  const int Bn = 2, S = 2048, D = 1024;
  const int M = Bn * S;                     // 4096
  const size_t WT  = (size_t)D * D;         // 1M elems
  const size_t QKV = (size_t)M * D;         // 4M elems

  // ws (bf16 elems): 4x W^T + Xbuf + q + k + vT + comb = 24M elems = 48 MB
  __bf16* WtQ  = (__bf16*)d_ws;
  __bf16* WtK  = WtQ + WT;
  __bf16* WtV  = WtK + WT;
  __bf16* WtO  = WtV + WT;
  __bf16* Xbuf = WtO + WT;
  __bf16* qws  = Xbuf + QKV;
  __bf16* kws  = qws + QKV;
  __bf16* vtw  = kws + QKV;
  __bf16* comb = vtw + QKV;

  dim3 blk(256);
  dim3 gridW((unsigned)((D * D) / 256));
  wt_convert_kernel<<<gridW, blk, 0, stream>>>(Wq, WtQ);
  wt_convert_kernel<<<gridW, blk, 0, stream>>>(Wk, WtK);
  wt_convert_kernel<<<gridW, blk, 0, stream>>>(Wv, WtV);
  wt_convert_kernel<<<gridW, blk, 0, stream>>>(Wo, WtO);

  dim3 gridX((unsigned)(QKV / 4 / 256));
  dim3 gridG(M / 128, D / 128);

  f32_to_bf16_kernel<<<gridX, blk, 0, stream>>>(Q, Xbuf);
  gemm128_wmma<0><<<gridG, blk, 0, stream>>>(Xbuf, WtQ, bq, qws);
  f32_to_bf16_kernel<<<gridX, blk, 0, stream>>>(K, Xbuf);
  gemm128_wmma<0><<<gridG, blk, 0, stream>>>(Xbuf, WtK, bk, kws);
  f32_to_bf16_kernel<<<gridX, blk, 0, stream>>>(V, Xbuf);
  gemm128_wmma<1><<<gridG, blk, 0, stream>>>(Xbuf, WtV, bv, vtw);

  dim3 gridA(S / 128, 16, Bn);
  attn_kernel<<<gridA, blk, 0, stream>>>(qws, kws, vtw, comb);

  gemm128_wmma<2><<<gridG, blk, 0, stream>>>(comb, WtO, bo, out);
}
